// MultiHeadAttention_10067403342152
// MI455X (gfx1250) — compile-verified, matrix operand order checked by reading, not run on hardware
//
#include <hip/hip_runtime.h>
#include <hip/hip_bf16.h>

#define D_MODEL  1024
#define N_HEADS  16
#define HEAD_DIM 64
#define BATCH    2
#define SEQ      2048
#define M_ROWS   (BATCH * SEQ)          // 4096

typedef __bf16 bf16;
typedef __attribute__((ext_vector_type(16))) __bf16 v16bf;
typedef __attribute__((ext_vector_type(8)))  __bf16 bf16x8;
typedef __attribute__((ext_vector_type(4)))  __bf16 bf16x4;
typedef __attribute__((ext_vector_type(8)))  float  v8f;

// ---------------------------------------------------------------------------
// WMMA helper: D = A(16x32 bf16) * B(32x16 bf16) + C(16x16 f32)
// ---------------------------------------------------------------------------
__device__ __forceinline__ v8f wmma_bf16(v16bf a, v16bf b, v8f c) {
    return __builtin_amdgcn_wmma_f32_16x16x32_bf16(
        /*neg_a=*/false, a, /*neg_b=*/false, b,
        /*c_mod=*/(short)0, c, /*reuse_a=*/false, /*reuse_b=*/false);
}

// ---------------------------------------------------------------------------
// Fragment loader for K-contiguous operand storage.
//   A: element (m,k) = base[m*ld + k]       (row-major activations)
//   B: element (k,n) = base[n*ld + k]       (transposed weights / K / V)
// Per documented bf16 layout each lane holds two contiguous 8-elem runs:
//   k = side*8 + [0..7]  and  k = 16 + side*8 + [0..7], side = lane>=16.
// => exactly two global_load_b128 per fragment.
// ---------------------------------------------------------------------------
__device__ __forceinline__ v16bf load_frag(const bf16* __restrict__ base,
                                           int ld, int lane) {
    int r  = lane & 15;
    int ks = (lane >> 4) << 3;                      // 0 or 8
    const bf16* p = base + (size_t)r * ld + ks;
    bf16x8 lo = *(const bf16x8*)(p);
    bf16x8 hi = *(const bf16x8*)(p + 16);
    v16bf f;
#pragma unroll
    for (int i = 0; i < 8; ++i) { f[i] = lo[i]; f[8 + i] = hi[i]; }
    return f;
}

// ---------------------------------------------------------------------------
// f32 -> bf16 conversion (vectorized 4 elems/thread)
// ---------------------------------------------------------------------------
__global__ void cvt_f32_to_bf16(const float* __restrict__ in,
                                bf16* __restrict__ out, int n4) {
    int i = blockIdx.x * blockDim.x + threadIdx.x;
    if (i < n4) {
        float4 x = ((const float4*)in)[i];
        bf16x4 y = { (bf16)x.x, (bf16)x.y, (bf16)x.z, (bf16)x.w };
        ((bf16x4*)out)[i] = y;
    }
}

// Transposed weight conversion: Wt[n][k] = (bf16)W[k][n]   (1024x1024)
__global__ void cvt_w_transpose(const float* __restrict__ W,
                                bf16* __restrict__ Wt) {
    int idx = blockIdx.x * blockDim.x + threadIdx.x;   // 0 .. 1M-1
    int n = idx >> 10;
    int k = idx & 1023;
    Wt[idx] = (bf16)W[k * D_MODEL + n];
}

// ---------------------------------------------------------------------------
// WMMA GEMM:  C[M,N] = A[M,K] * Bt[N,K]^T + bias
// One wave computes a 16x64 strip (4 accumulators, A-fragment reused 4x).
//   mode 0: bf16 epilogue into head layout      [B, H, S, Dh]   (Q, K)
//   mode 2: bf16 epilogue into transposed heads [B, H, Dh, S]   (V)
//   mode 1: f32 epilogue row-major [M, N]                        (out)
// ---------------------------------------------------------------------------
__global__ void gemm_wmma(const bf16* __restrict__ A,
                          const bf16* __restrict__ Bt,
                          const float* __restrict__ bias,
                          void* __restrict__ Cout,
                          int M, int N, int K, int mode) {
    int wave = blockIdx.x * (blockDim.x >> 5) + (threadIdx.x >> 5);
    int lane = threadIdx.x & 31;
    int n64  = N >> 6;                                 // 64-wide strips
    int tile_m = wave / n64;
    int tile_n = wave - tile_m * n64;                  // strip index

    const bf16* Ab = A  + (size_t)tile_m * 16 * K;
    const bf16* Bb = Bt + (size_t)tile_n * 64 * K;

    v8f acc0 = {}, acc1 = {}, acc2 = {}, acc3 = {};
    for (int k = 0; k < K; k += 32) {
        v16bf af = load_frag(Ab + k, K, lane);
        acc0 = wmma_bf16(af, load_frag(Bb + k,                     K, lane), acc0);
        acc1 = wmma_bf16(af, load_frag(Bb + (size_t)16 * K + k,    K, lane), acc1);
        acc2 = wmma_bf16(af, load_frag(Bb + (size_t)32 * K + k,    K, lane), acc2);
        acc3 = wmma_bf16(af, load_frag(Bb + (size_t)48 * K + k,    K, lane), acc3);
    }

    v8f accs[4] = { acc0, acc1, acc2, acc3 };
    int n_local = lane & 15;
#pragma unroll
    for (int j = 0; j < 4; ++j) {
        int gn = tile_n * 64 + j * 16 + n_local;
        float bv = bias[gn];
#pragma unroll
        for (int v = 0; v < 8; ++v) {
            int gm = tile_m * 16 + v + ((lane >> 4) << 3);
            float val = accs[j][v] + bv;
            if (mode == 0) {                            // [B,H,S,Dh] bf16
                int b = gm / SEQ, s = gm - (gm / SEQ) * SEQ;
                int h = gn / HEAD_DIM, d = gn & (HEAD_DIM - 1);
                ((bf16*)Cout)[(((size_t)b * N_HEADS + h) * SEQ + s) * HEAD_DIM + d] =
                    (bf16)val;
            } else if (mode == 2) {                     // [B,H,Dh,S] bf16
                int b = gm / SEQ, s = gm - (gm / SEQ) * SEQ;
                int h = gn / HEAD_DIM, d = gn & (HEAD_DIM - 1);
                ((bf16*)Cout)[(((size_t)b * N_HEADS + h) * HEAD_DIM + d) * SEQ + s] =
                    (bf16)val;
            } else {                                    // f32 row-major
                ((float*)Cout)[(size_t)gm * N + gn] = val;
            }
        }
    }
}

// ---------------------------------------------------------------------------
// Fused causal attention per (b, h, 16-query-row tile).
// Block = 256 threads (8 waves). LDS: scores 16 x 2052 f32 + partial 16 x 64.
// ---------------------------------------------------------------------------
#define SC_STRIDE (SEQ + 4)                               // 2052 floats
#define ATTN_LDS_BYTES ((16 * SC_STRIDE + 16 * HEAD_DIM) * (int)sizeof(float))

__global__ void attn_fused(const bf16* __restrict__ Q,
                           const bf16* __restrict__ Kh,
                           const bf16* __restrict__ Vt,
                           float* __restrict__ Wout,
                           bf16* __restrict__ attn_out) {
    extern __shared__ float smem[];
    float* scores  = smem;                                 // [16][SC_STRIDE]
    float* partial = smem + 16 * SC_STRIDE;                // [16][64]

    int bh    = blockIdx.y;                                // b*16 + h
    int qtile = blockIdx.x;                                // 0..127
    int b = bh / N_HEADS, h = bh - b * N_HEADS;
    int wave = threadIdx.x >> 5;
    int lane = threadIdx.x & 31;
    int q0 = qtile * 16;

    const bf16* Qb = Q  + ((size_t)bh * SEQ + q0) * HEAD_DIM;
    const bf16* Kb = Kh + (size_t)bh * SEQ * HEAD_DIM;     // [S][Dh]
    const bf16* Vb = Vt + (size_t)bh * HEAD_DIM * SEQ;     // [Dh][S]

    // Q tile fragments (16x64 = two k-steps), loaded once per wave.
    v16bf qa0 = load_frag(Qb,      HEAD_DIM, lane);
    v16bf qa1 = load_frag(Qb + 32, HEAD_DIM, lane);

    const float scale = 0.125f;                            // 1/sqrt(64)

    // ---- Phase 1: scores = scale * Q K^T with causal mask -> LDS ----------
    for (int jt = wave; jt < (SEQ >> 4); jt += 8) {
        const bf16* Kt = Kb + (size_t)jt * 16 * HEAD_DIM;  // 16 rows of K
        v8f s = {};
        s = wmma_bf16(qa0, load_frag(Kt,      HEAD_DIM, lane), s);
        s = wmma_bf16(qa1, load_frag(Kt + 32, HEAD_DIM, lane), s);
        int n_local = lane & 15;
        int kglob = jt * 16 + n_local;
#pragma unroll
        for (int v = 0; v < 8; ++v) {
            int m = v + ((lane >> 4) << 3);
            int qglob = q0 + m;
            float val = (qglob >= kglob) ? s[v] * scale : -10000.0f;
            scores[m * SC_STRIDE + kglob] = val;
        }
    }
    __syncthreads();

    // ---- Phase 2: row softmax in LDS (float4), write w to global ----------
    {
        int r   = threadIdx.x >> 4;                        // 0..15
        int sub = threadIdx.x & 15;
        float* row = scores + r * SC_STRIDE;

        float mx = -1e30f;
        for (int c = sub * 4; c < SEQ; c += 64) {
            float4 x = *(const float4*)(row + c);
            mx = fmaxf(mx, fmaxf(fmaxf(x.x, x.y), fmaxf(x.z, x.w)));
        }
#pragma unroll
        for (int off = 8; off; off >>= 1) mx = fmaxf(mx, __shfl_xor(mx, off, 16));

        float sum = 0.0f;
        for (int c = sub * 4; c < SEQ; c += 64) {
            float4 x = *(const float4*)(row + c);
            x.x = __expf(x.x - mx); x.y = __expf(x.y - mx);
            x.z = __expf(x.z - mx); x.w = __expf(x.w - mx);
            *(float4*)(row + c) = x;
            sum += x.x + x.y + x.z + x.w;
        }
#pragma unroll
        for (int off = 8; off; off >>= 1) sum += __shfl_xor(sum, off, 16);

        float inv = 1.0f / sum;
        float* wrow = Wout + ((size_t)bh * SEQ + q0 + r) * SEQ;
        for (int c = sub * 4; c < SEQ; c += 64) {
            float4 x = *(const float4*)(row + c);
            x.x *= inv; x.y *= inv; x.z *= inv; x.w *= inv;
            *(float4*)(row + c) = x;
            *(float4*)(wrow + c) = x;                      // w output (HBM)
        }
    }
    __syncthreads();

    // ---- Phase 3: attn = P @ V (16 x 64): 4 n-tiles x 2 k-halves ----------
    int ntile = wave & 3;
    int khalf = wave >> 2;
    int kbeg = khalf * (SEQ / 2);
    int m  = lane & 15;
    int ks = (lane >> 4) << 3;
    const float* rowp = scores + m * SC_STRIDE;

    v8f acc = {};
    for (int k = kbeg; k < kbeg + SEQ / 2; k += 32) {
        // A fragment from LDS (float4 reads), f32 -> bf16 in registers
        const float* p = rowp + k + ks;
        float4 a0 = *(const float4*)(p);
        float4 a1 = *(const float4*)(p + 4);
        float4 a2 = *(const float4*)(p + 16);
        float4 a3 = *(const float4*)(p + 20);
        v16bf a;
        a[0]  = (bf16)a0.x; a[1]  = (bf16)a0.y; a[2]  = (bf16)a0.z; a[3]  = (bf16)a0.w;
        a[4]  = (bf16)a1.x; a[5]  = (bf16)a1.y; a[6]  = (bf16)a1.z; a[7]  = (bf16)a1.w;
        a[8]  = (bf16)a2.x; a[9]  = (bf16)a2.y; a[10] = (bf16)a2.z; a[11] = (bf16)a2.w;
        a[12] = (bf16)a3.x; a[13] = (bf16)a3.y; a[14] = (bf16)a3.z; a[15] = (bf16)a3.w;

        // B fragment from V^T [Dh][S]: K-contiguous -> 2x b128
        v16bf bfr = load_frag(Vb + (size_t)(ntile * 16) * SEQ + k, SEQ, lane);
        acc = wmma_bf16(a, bfr, acc);
    }

    int n_local = lane & 15;
    if (khalf == 1) {
#pragma unroll
        for (int v = 0; v < 8; ++v) {
            int mm = v + ((lane >> 4) << 3);
            partial[mm * HEAD_DIM + ntile * 16 + n_local] = acc[v];
        }
    }
    __syncthreads();
    if (khalf == 0) {
#pragma unroll
        for (int v = 0; v < 8; ++v) {
            int mm = v + ((lane >> 4) << 3);
            float val = acc[v] + partial[mm * HEAD_DIM + ntile * 16 + n_local];
            int s_glob = q0 + mm;
            attn_out[((size_t)b * SEQ + s_glob) * D_MODEL +
                     h * HEAD_DIM + ntile * 16 + n_local] = (bf16)val;
        }
    }
}

// ---------------------------------------------------------------------------
// Host-side orchestration
// ---------------------------------------------------------------------------
extern "C" void kernel_launch(void* const* d_in, const int* in_sizes, int n_in,
                              void* d_out, int out_size, void* d_ws, size_t ws_size,
                              hipStream_t stream) {
    (void)in_sizes; (void)n_in; (void)out_size; (void)ws_size;

    const float* value  = (const float*)d_in[0];
    const float* key_in = (const float*)d_in[1];
    const float* query  = (const float*)d_in[2];
    const float* Wq = (const float*)d_in[3];  const float* bq = (const float*)d_in[4];
    const float* Wk = (const float*)d_in[5];  const float* bk = (const float*)d_in[6];
    const float* Wv = (const float*)d_in[7];  const float* bv = (const float*)d_in[8];
    const float* Wo = (const float*)d_in[9];  const float* bo = (const float*)d_in[10];

    // d_out = [ out (B*S*D f32) | w (B*H*S*S f32) ]
    float* out_p = (float*)d_out;
    float* w_p   = out_p + (size_t)BATCH * SEQ * D_MODEL;

    // Workspace carve-up (bf16 buffers), 64 MB total.
    char* ws = (char*)d_ws;
    const size_t WSZ = (size_t)D_MODEL * D_MODEL * sizeof(bf16);   // 2 MB
    const size_t XSZ = (size_t)M_ROWS * D_MODEL * sizeof(bf16);    // 8 MB
    bf16* WqT = (bf16*)(ws);
    bf16* WkT = (bf16*)(ws + WSZ);
    bf16* WvT = (bf16*)(ws + 2 * WSZ);
    bf16* WoT = (bf16*)(ws + 3 * WSZ);
    bf16* Xq  = (bf16*)(ws + 4 * WSZ);
    bf16* Xk  = (bf16*)(ws + 4 * WSZ + XSZ);
    bf16* Xv  = (bf16*)(ws + 4 * WSZ + 2 * XSZ);
    bf16* Qh  = (bf16*)(ws + 4 * WSZ + 3 * XSZ);
    bf16* Kh  = (bf16*)(ws + 4 * WSZ + 4 * XSZ);
    bf16* Vt  = (bf16*)(ws + 4 * WSZ + 5 * XSZ);                   // [B,H,Dh,S]
    bf16* At  = (bf16*)(ws + 4 * WSZ + 6 * XSZ);                   // attn concat

    const int T = 256;
    const int n_act = M_ROWS * D_MODEL;                            // 4,194,304

    // 1) Precision conversion (activations + transposed weights).
    cvt_f32_to_bf16<<<(n_act / 4) / T, T, 0, stream>>>(query,  Xq, n_act / 4);
    cvt_f32_to_bf16<<<(n_act / 4) / T, T, 0, stream>>>(key_in, Xk, n_act / 4);
    cvt_f32_to_bf16<<<(n_act / 4) / T, T, 0, stream>>>(value,  Xv, n_act / 4);
    cvt_w_transpose<<<(D_MODEL * D_MODEL) / T, T, 0, stream>>>(Wq, WqT);
    cvt_w_transpose<<<(D_MODEL * D_MODEL) / T, T, 0, stream>>>(Wk, WkT);
    cvt_w_transpose<<<(D_MODEL * D_MODEL) / T, T, 0, stream>>>(Wv, WvT);
    cvt_w_transpose<<<(D_MODEL * D_MODEL) / T, T, 0, stream>>>(Wo, WoT);

    // 2) Q/K/V projections: [4096,1024] x [1024,1024], 16x64 strip per wave.
    int gemm_blocks = (M_ROWS / 16) * (D_MODEL / 64) / 8;          // 512
    gemm_wmma<<<gemm_blocks, T, 0, stream>>>(Xq, WqT, bq, Qh,
                                             M_ROWS, D_MODEL, D_MODEL, 0);
    gemm_wmma<<<gemm_blocks, T, 0, stream>>>(Xk, WkT, bk, Kh,
                                             M_ROWS, D_MODEL, D_MODEL, 0);
    gemm_wmma<<<gemm_blocks, T, 0, stream>>>(Xv, WvT, bv, Vt,
                                             M_ROWS, D_MODEL, D_MODEL, 2);

    // 3) Fused attention: grid (qtiles, B*H), ~135 KB dynamic LDS.
    dim3 agrid(SEQ / 16, BATCH * N_HEADS);
    attn_fused<<<agrid, T, ATTN_LDS_BYTES, stream>>>(Qh, Kh, Vt, w_p, At);

    // 4) Output projection: f32 epilogue straight into d_out.
    gemm_wmma<<<gemm_blocks, T, 0, stream>>>(At, WoT, bo, out_p,
                                             M_ROWS, D_MODEL, D_MODEL, 1);
}
